// SyllableLSTM_59021440581942
// MI455X (gfx1250) — matrix operand order repaired and checked by reading, not compiled, not run on hardware
//
#include <hip/hip_runtime.h>

// ---------------------------------------------------------------------------
// SyllableLSTM on gfx1250 (MI455X):
//   Kernel 0a/0b: one-time f32->bf16 prepack of spectrogram / W_ih into
//                 native WMMA fragment layouts (A: 16x32, B: 32x16).
//   Kernel 1:     x_proj[T,256] = spec.T @ W_ih.T + (b_ih+b_hh) via
//                 v_wmma_f32_16x16x32_bf16, pure b128 loads in the K loop.
//   Kernel 2:     sequential LSTM recurrence (1 workgroup) + log_softmax.
// ---------------------------------------------------------------------------

#define T_STEPS 32768
#define IN_DIM  1024
#define HID     64
#define GATES   256            // 4*HID
#define TT_TILES (T_STEPS / 16)   // 2048
#define KT_TILES (IN_DIM / 32)    // 32
#define JT_TILES (GATES / 16)     // 16
#define FRAG_DW  256              // dwords per 16x32 (or 32x16) bf16 tile

typedef __bf16 bf16x16 __attribute__((ext_vector_type(16)));
typedef float  f32x8   __attribute__((ext_vector_type(8)));

union BF16Frag { unsigned int u[8]; uint4 q[2]; bf16x16 v; };

// round-to-nearest-even f32 -> bf16, packed pair (lo in [15:0], hi in [31:16])
__device__ __forceinline__ unsigned int pack2_bf16(float lo, float hi) {
  union { float f; unsigned int u; } a, b;
  a.f = lo; b.f = hi;
  unsigned int ra = (a.u + 0x7FFFu + ((a.u >> 16) & 1u)) >> 16;
  unsigned int rb = (b.u + 0x7FFFu + ((b.u >> 16) & 1u)) >> 16;
  return (ra & 0xFFFFu) | (rb << 16);
}

// ---------------------------------------------------------------------------
// Kernel 0a: prepack spectrogram into A-fragment layout (dense 16-bit A,
// ISA 7.12.2):  lane l: m=l&15, hi=l>>4; VGPR v holds K pair
//   k = ((v<4) ? 2v : 8+2v) + 8*hi.
// Tile (tt, kt) stored at apack[(tt*32+kt)*256 + l*8 + v].
// Grid: 2048 blocks x 256 thr; block tt loops over kt.
// ---------------------------------------------------------------------------
__global__ __launch_bounds__(256)
void prepack_A(const float* __restrict__ spec, unsigned int* __restrict__ apack) {
  const int tt = blockIdx.x;
  const int l  = threadIdx.x >> 3;
  const int v  = threadIdx.x & 7;
  const int m  = l & 15;
  const int hi = l >> 4;
  const int t  = tt * 16 + m;
  const int kk = ((v < 4) ? (2 * v) : (8 + 2 * v)) + hi * 8;
  for (int kt = 0; kt < KT_TILES; ++kt) {
    const int k = kt * 32 + kk;
    const float x0 = spec[(size_t)k * T_STEPS + t];
    const float x1 = spec[(size_t)(k + 1) * T_STEPS + t];
    apack[(size_t)(tt * KT_TILES + kt) * FRAG_DW + threadIdx.x] = pack2_bf16(x0, x1);
  }
}

// ---------------------------------------------------------------------------
// Kernel 0b: prepack W_ih into B-fragment layout (16-bit B, ISA 7.12.4
// scaled to K=32): lane l: n=l&15, hi=l>>4; VGPR v holds K pair
//   k = ((v<4) ? 2v : 8+2v) + 8*hi.
// Tile (kt, jt) stored at bpack[(kt*16+jt)*256 + l*8 + v].
// Grid: 512 blocks x 256 thr (one tile per block).
// ---------------------------------------------------------------------------
__global__ __launch_bounds__(256)
void prepack_B(const float* __restrict__ Wih, unsigned int* __restrict__ bpack) {
  const int ti = blockIdx.x;            // kt*16 + jt
  const int kt = ti >> 4;
  const int jt = ti & 15;
  const int l  = threadIdx.x >> 3;
  const int v  = threadIdx.x & 7;
  const int n  = l & 15;
  const int hi = l >> 4;
  const int j  = jt * 16 + n;
  const int k  = kt * 32 + ((v < 4) ? (2 * v) : (8 + 2 * v)) + hi * 8;
  const size_t idx = (size_t)j * IN_DIM + k;
  bpack[(size_t)ti * FRAG_DW + threadIdx.x] = pack2_bf16(Wih[idx], Wih[idx + 1]);
}

// ---------------------------------------------------------------------------
// Kernel 1 (packed): GEMM with prepacked bf16 fragments.
// Block = 128 threads = 4 waves, one t-tile per block.
// Wave w owns j-tiles 4w..4w+3 (4 accumulators); one A frag feeds 4 WMMAs.
// Inner loop: 2 b128 loads (A) + 8 b128 loads (B) + 4 WMMA.
// ---------------------------------------------------------------------------
__global__ __launch_bounds__(128)
void xproj_gemm_wmma_packed(const unsigned int* __restrict__ apack,
                            const unsigned int* __restrict__ bpack,
                            const float* __restrict__ bih,
                            const float* __restrict__ bhh,
                            float* __restrict__ xp) {
  const int tt   = blockIdx.x;
  const int wave = threadIdx.x >> 5;    // 0..3
  const int lane = threadIdx.x & 31;
  const int n    = lane & 15;
  const int hi   = lane >> 4;

  f32x8 acc[4] = {};

  const unsigned int* aptr = apack + (size_t)tt * KT_TILES * FRAG_DW + lane * 8;
  for (int kt = 0; kt < KT_TILES; ++kt) {
    BF16Frag a;
    a.q[0] = *(const uint4*)(aptr);
    a.q[1] = *(const uint4*)(aptr + 4);
    aptr += FRAG_DW;

    BF16Frag b[4];
#pragma unroll
    for (int q = 0; q < 4; ++q) {
      const unsigned int* bptr =
          bpack + (size_t)(kt * JT_TILES + wave * 4 + q) * FRAG_DW + lane * 8;
      b[q].q[0] = *(const uint4*)(bptr);
      b[q].q[1] = *(const uint4*)(bptr + 4);
    }
#pragma unroll
    for (int q = 0; q < 4; ++q)
      acc[q] = __builtin_amdgcn_wmma_f32_16x16x32_bf16(
          false, a.v, false, b[q].v, (short)0, acc[q], false, false);
  }

  // epilogue: + (b_ih + b_hh); D layout: VGPR r -> M = r + 8*hi, N = lane&15
#pragma unroll
  for (int q = 0; q < 4; ++q) {
    const int j = (wave * 4 + q) * 16 + n;
    const float bias = bih[j] + bhh[j];
#pragma unroll
    for (int r = 0; r < 8; ++r) {
      const int trow = tt * 16 + r + hi * 8;
      xp[(size_t)trow * GATES + j] = acc[q][r] + bias;
    }
  }
}

// ---------------------------------------------------------------------------
// Kernel 1 (fallback, convert-in-loop) — used when workspace can't hold the
// packed operands. Same math as round 1.
// ---------------------------------------------------------------------------
__global__ __launch_bounds__(256)
void xproj_gemm_wmma_direct(const float* __restrict__ spec,
                            const float* __restrict__ Wih,
                            const float* __restrict__ bih,
                            const float* __restrict__ bhh,
                            float* __restrict__ xp) {
  const int tt   = blockIdx.x;
  const int wave = threadIdx.x >> 5;
  const int lane = threadIdx.x & 31;
  const int m    = lane & 15;
  const int hi   = lane >> 4;
  const int t    = tt * 16 + m;
  const int jA   = wave * 32 + m;
  const int jB   = jA + 16;

  f32x8 acc0 = {}, acc1 = {};
  for (int k0 = 0; k0 < IN_DIM; k0 += 32) {
    BF16Frag a;
#pragma unroll
    for (int v = 0; v < 8; ++v) {
      const int k = ((v < 4) ? (2 * v) : (8 + 2 * v)) + hi * 8;
      a.u[v] = pack2_bf16(spec[(size_t)(k0 + k) * T_STEPS + t],
                          spec[(size_t)(k0 + k + 1) * T_STEPS + t]);
    }
    BF16Frag b0, b1;
#pragma unroll
    for (int v = 0; v < 8; ++v) {
      const int k = ((v < 4) ? (2 * v) : (8 + 2 * v)) + hi * 8;
      const size_t iA = (size_t)jA * IN_DIM + (k0 + k);
      const size_t iB = (size_t)jB * IN_DIM + (k0 + k);
      b0.u[v] = pack2_bf16(Wih[iA], Wih[iA + 1]);
      b1.u[v] = pack2_bf16(Wih[iB], Wih[iB + 1]);
    }
    acc0 = __builtin_amdgcn_wmma_f32_16x16x32_bf16(false, a.v, false, b0.v,
                                                   (short)0, acc0, false, false);
    acc1 = __builtin_amdgcn_wmma_f32_16x16x32_bf16(false, a.v, false, b1.v,
                                                   (short)0, acc1, false, false);
  }
  const float biasA = bih[jA] + bhh[jA];
  const float biasB = bih[jB] + bhh[jB];
#pragma unroll
  for (int r = 0; r < 8; ++r) {
    const int trow = tt * 16 + r + hi * 8;
    xp[(size_t)trow * GATES + jA] = acc0[r] + biasA;
    xp[(size_t)trow * GATES + jB] = acc1[r] + biasB;
  }
}

// ---------------------------------------------------------------------------
// Kernel 2: sequential recurrence. One workgroup, 256 threads.
// ---------------------------------------------------------------------------
__device__ __forceinline__ float sigm(float x) {
  return 1.0f / (1.0f + __expf(-x));
}
__device__ __forceinline__ float fast_tanh(float x) {
  return 1.0f - 2.0f / (__expf(2.0f * x) + 1.0f);
}

__global__ __launch_bounds__(256)
void lstm_recurrence(const float* __restrict__ xp,
                     const float* __restrict__ Whh,
                     float* __restrict__ out) {
  __shared__ float h_lds[HID];
  __shared__ float gate_lds[GATES];
  __shared__ float red[2];

  const int j = threadIdx.x;

  float w[HID];
#pragma unroll
  for (int k = 0; k < HID; ++k) w[k] = Whh[(size_t)j * HID + k];

  float c = 0.0f;
  if (j < HID) h_lds[j] = 0.0f;
  __syncthreads();

  for (int t = 0; t < T_STEPS; ++t) {
    if (j == 0 && (t + 16) < T_STEPS)
      __builtin_prefetch(xp + (size_t)(t + 16) * GATES, 0, 1);

    float g = xp[(size_t)t * GATES + j];
#pragma unroll
    for (int k = 0; k < HID; ++k) g = fmaf(w[k], h_lds[k], g);
    gate_lds[j] = g;
    __syncthreads();

    if (j < HID) {
      const float ig = sigm(gate_lds[j]);
      const float fg = sigm(gate_lds[HID + j]);
      const float gg = fast_tanh(gate_lds[2 * HID + j]);
      const float og = sigm(gate_lds[3 * HID + j]);
      c = fg * c + ig * gg;
      h_lds[j] = og * fast_tanh(c);
    }
    __syncthreads();
  }

  if (j == 0) {
    float mx = h_lds[0];
    for (int k = 1; k < HID; ++k) mx = fmaxf(mx, h_lds[k]);
    float s = 0.0f;
    for (int k = 0; k < HID; ++k) s += __expf(h_lds[k] - mx);
    red[0] = mx;
    red[1] = logf(s);
  }
  __syncthreads();
  if (j < HID) out[j] = h_lds[j] - red[0] - red[1];
}

// ---------------------------------------------------------------------------
extern "C" void kernel_launch(void* const* d_in, const int* in_sizes, int n_in,
                              void* d_out, int out_size, void* d_ws, size_t ws_size,
                              hipStream_t stream) {
  (void)in_sizes; (void)n_in; (void)out_size;
  const float* spec = (const float*)d_in[0];   // (1024, 32768)
  const float* Wih  = (const float*)d_in[1];   // (256, 1024)
  const float* Whh  = (const float*)d_in[2];   // (256, 64)
  const float* bih  = (const float*)d_in[3];   // (256,)
  const float* bhh  = (const float*)d_in[4];   // (256,)
  float* out = (float*)d_out;                  // (64,)

  const size_t xp_bytes    = (size_t)T_STEPS * GATES * sizeof(float);          // 32 MB
  const size_t apack_dw    = (size_t)TT_TILES * KT_TILES * FRAG_DW;            // 16.7M dw
  const size_t bpack_dw    = (size_t)KT_TILES * JT_TILES * FRAG_DW;            // 131K dw
  const size_t need        = xp_bytes + (apack_dw + bpack_dw) * 4;

  float* xp = (float*)d_ws;

  if (ws_size >= need) {
    unsigned int* apack = (unsigned int*)((char*)d_ws + xp_bytes);
    unsigned int* bpack = apack + apack_dw;
    prepack_A<<<TT_TILES, 256, 0, stream>>>(spec, apack);
    prepack_B<<<KT_TILES * JT_TILES, 256, 0, stream>>>(Wih, bpack);
    xproj_gemm_wmma_packed<<<TT_TILES, 128, 0, stream>>>(apack, bpack, bih, bhh, xp);
  } else {
    xproj_gemm_wmma_direct<<<TT_TILES, 256, 0, stream>>>(spec, Wih, bih, bhh, xp);
  }
  lstm_recurrence<<<1, 256, 0, stream>>>(xp, Whh, out);
}
